// TopoGCN_71829033058960
// MI455X (gfx1250) — matrix operand order repaired and checked by reading, not compiled, not run on hardware
//
#include <hip/hip_runtime.h>

#define N_NODES 10000
#define N_EDGES 160000
#define DIM 512
#define HID 256
#define EDIM 16

typedef __attribute__((ext_vector_type(16))) __bf16 v16bf;
typedef __attribute__((ext_vector_type(8)))  float  v8f;

union FragBF { v16bf v; uint4 q[2]; };

// ---------------- weight transpose + fp32->bf16 convert ----------------
// dst[n*K + k] = src[(rowoff + k)*ld + n]   (dst is [Nout][K] bf16)
__global__ void transcvt_kernel(const float* __restrict__ src, __bf16* __restrict__ dst,
                                int K, int Nout, int ld, int rowoff) {
  int i = blockIdx.x * blockDim.x + threadIdx.x;
  if (i >= K * Nout) return;
  int n = i / K, k = i - n * K;
  dst[i] = (__bf16)src[(size_t)(rowoff + k) * ld + n];
}

// ---------------- WMMA bf16 GEMM: C[M,Nout] = A[M,K](f32) @ Bt^T + bias ----
// Bt is [Nout][K] bf16 (row n = column n of W). One wave per 16x16 C tile.
// A-rows >= M are clamped to row M-1 (their products only land in masked-out
// C rows), so the inner loop is branch-free vector loads.
__global__ void __launch_bounds__(256)
wmma_gemm_kernel(const float* __restrict__ A, int lda,
                 const __bf16* __restrict__ Bt,
                 float* __restrict__ C,
                 const float* __restrict__ bias,
                 int M, int K, int Nout) {
  const int lane = threadIdx.x & 31;
  const int wid  = (blockIdx.x * blockDim.x + threadIdx.x) >> 5;
  const int tilesN = Nout >> 4;
  const int tilesM = (M + 15) >> 4;
  if (wid >= tilesM * tilesN) return;
  const int tm = wid / tilesN;
  const int tn = wid - tm * tilesN;
  const int h = lane >> 4;      // half-wave select
  const int r = lane & 15;
  int row = tm * 16 + r;        // A row held by this lane
  if (row >= M) row = M - 1;    // clamp: contributes only to masked C rows
  // A 16-bit 16x32 layout: lane holds row r, K = k0 + 16h + {0..15}, contiguous
  const float*  __restrict__ arow = A  + (size_t)row * lda + 16 * h;
  // B 32x16 layout: lane holds col r, K = k0 + 16h + {0..15}, contiguous
  const __bf16* __restrict__ brow = Bt + (size_t)(tn * 16 + r) * K + 16 * h;
  v8f acc = {0.f, 0.f, 0.f, 0.f, 0.f, 0.f, 0.f, 0.f};
#pragma unroll 2
  for (int k0 = 0; k0 < K; k0 += 32) {
    FragBF fa, fb;
    const float4 x0 = *(const float4*)(arow + k0);
    const float4 x1 = *(const float4*)(arow + k0 + 4);
    const float4 y0 = *(const float4*)(arow + k0 + 8);
    const float4 y1 = *(const float4*)(arow + k0 + 12);
    fa.v[0]  = (__bf16)x0.x; fa.v[1]  = (__bf16)x0.y;
    fa.v[2]  = (__bf16)x0.z; fa.v[3]  = (__bf16)x0.w;
    fa.v[4]  = (__bf16)x1.x; fa.v[5]  = (__bf16)x1.y;
    fa.v[6]  = (__bf16)x1.z; fa.v[7]  = (__bf16)x1.w;
    fa.v[8]  = (__bf16)y0.x; fa.v[9]  = (__bf16)y0.y;
    fa.v[10] = (__bf16)y0.z; fa.v[11] = (__bf16)y0.w;
    fa.v[12] = (__bf16)y1.x; fa.v[13] = (__bf16)y1.y;
    fa.v[14] = (__bf16)y1.z; fa.v[15] = (__bf16)y1.w;
    const uint4* bp = (const uint4*)(brow + k0);
    fb.q[0] = bp[0];
    fb.q[1] = bp[1];
    acc = __builtin_amdgcn_wmma_f32_16x16x32_bf16(false, fa.v, false, fb.v,
                                                  (short)0, acc, false, false);
  }
  const float bb = bias ? bias[tn * 16 + r] : 0.f;
#pragma unroll
  for (int v = 0; v < 8; ++v) {            // D: VGPR v -> row v + 8h, col r
    int orow = tm * 16 + v + 8 * h;
    if (orow < M) C[(size_t)orow * Nout + tn * 16 + r] = acc[v] + bb;
  }
}

// ---------------- CGConv edge phase (bandwidth-bound) ----------------
__device__ __forceinline__ float sigmoidf_(float x) { return 1.f / (1.f + __expf(-x)); }
__device__ __forceinline__ float softplusf_(float x) {
  return x > 15.f ? x : __logf(1.f + __expf(x));
}

__global__ void __launch_bounds__(256)
cg_edge_kernel(const int* __restrict__ ei,
               const float* __restrict__ Af, const float* __restrict__ Bf,
               const float* __restrict__ As, const float* __restrict__ Bs,
               const float* __restrict__ eattr,
               const float* __restrict__ WfE, const float* __restrict__ WsE,
               float* __restrict__ aggr, int nE) {
  __shared__ float sWf[EDIM * DIM];   // 32 KB
  __shared__ float sWs[EDIM * DIM];   // 32 KB
  for (int i = threadIdx.x; i < EDIM * DIM; i += blockDim.x) {
    sWf[i] = WfE[i];
    sWs[i] = WsE[i];
  }
  __syncthreads();
  const int lane   = threadIdx.x & 31;
  const int wave   = (blockIdx.x * blockDim.x + threadIdx.x) >> 5;
  const int nwaves = (gridDim.x * blockDim.x) >> 5;
  for (int e = wave; e < nE; e += nwaves) {
    const int s = ei[e];
    const int d = ei[nE + e];
    float ev = (lane < EDIM) ? eattr[(size_t)e * EDIM + lane] : 0.f;
    float ej[EDIM];
#pragma unroll
    for (int j = 0; j < EDIM; ++j) ej[j] = __shfl(ev, j);
    const float* afp = Af + (size_t)d * DIM;
    const float* bfp = Bf + (size_t)s * DIM;
    const float* asp = As + (size_t)d * DIM;
    const float* bsp = Bs + (size_t)s * DIM;
    float* agp = aggr + (size_t)d * DIM;
#pragma unroll 4
    for (int it = 0; it < DIM / 32; ++it) {
      const int c = lane + 32 * it;
      float F = afp[c] + bfp[c];   // bias folded into Af/As by GEMM
      float S = asp[c] + bsp[c];
#pragma unroll
      for (int j = 0; j < EDIM; ++j) {
        F = fmaf(ej[j], sWf[j * DIM + c], F);
        S = fmaf(ej[j], sWs[j * DIM + c], S);
      }
      atomicAdd(&agp[c], sigmoidf_(F) * softplusf_(S));
    }
  }
}

// ---------------- BN(eval) + residual + ReLU ----------------
__global__ void cg_epilogue_kernel(const float* __restrict__ xin, const float* __restrict__ aggr,
                                   const float* __restrict__ gamma, const float* __restrict__ beta,
                                   const float* __restrict__ mean, const float* __restrict__ var,
                                   float* __restrict__ xout, int total) {
  int i = blockIdx.x * blockDim.x + threadIdx.x;
  if (i >= total) return;
  int c = i & (DIM - 1);
  float bn = (aggr[i] - mean[c]) * __frsqrt_rn(var[c] + 1e-5f) * gamma[c] + beta[c];
  xout[i] = fmaxf(xin[i] + bn, 0.f);
}

// ---------------- degree helpers ----------------
__global__ void fill_kernel(float* __restrict__ p, float v, int n) {
  int i = blockIdx.x * blockDim.x + threadIdx.x;
  if (i < n) p[i] = v;
}
__global__ void deg_kernel(const int* __restrict__ dst, float* __restrict__ deg, int nE) {
  int i = blockIdx.x * blockDim.x + threadIdx.x;
  if (i < nE) atomicAdd(&deg[dst[i]], 1.f);
}
__global__ void rsqrt_kernel(float* __restrict__ p, int n) {
  int i = blockIdx.x * blockDim.x + threadIdx.x;
  if (i < n) p[i] = __frsqrt_rn(p[i]);
}

// ---------------- GCN scatter + epilogue ----------------
__global__ void __launch_bounds__(256)
gcn_scatter_kernel(const int* __restrict__ ei, const float* __restrict__ dinv,
                   const float* __restrict__ h, float* __restrict__ out, int nE, int C) {
  const int lane   = threadIdx.x & 31;
  const int wave   = (blockIdx.x * blockDim.x + threadIdx.x) >> 5;
  const int nwaves = (gridDim.x * blockDim.x) >> 5;
  for (int e = wave; e < nE; e += nwaves) {
    int s = ei[e], d = ei[nE + e];
    float coef = dinv[s] * dinv[d];
    const float* hp = h + (size_t)s * C;
    float* op = out + (size_t)d * C;
    for (int c = lane; c < C; c += 32) atomicAdd(&op[c], coef * hp[c]);
  }
}

__global__ void gcn_epilogue_kernel(float* __restrict__ out, const float* __restrict__ h,
                                    const float* __restrict__ dinv, const float* __restrict__ bias,
                                    int n, int C) {
  int i = blockIdx.x * blockDim.x + threadIdx.x;
  if (i >= n * C) return;
  int node = i / C, c = i - node * C;
  float di = dinv[node];
  out[i] = fmaxf(out[i] + di * di * h[i] + bias[c], 0.f);
}

// ---------------- dense head ----------------
__global__ void gterm_kernel(const float* __restrict__ g, const float* __restrict__ d1W,
                             const float* __restrict__ d1b, float* __restrict__ gt) {
  int d = threadIdx.x;  // 512 threads
  float acc = d1b[d];
  for (int j = 0; j < DIM; ++j) acc = fmaf(g[j], d1W[(size_t)(HID + j) * DIM + d], acc);
  gt[d] = acc;
}

__global__ void __launch_bounds__(256)
final_kernel(const float* __restrict__ h1, const float* __restrict__ gt,
             const float* __restrict__ d2w, const float* __restrict__ d2b,
             float* __restrict__ out, int n) {
  const int lane = threadIdx.x & 31;
  const int node = (blockIdx.x * blockDim.x + threadIdx.x) >> 5;
  if (node >= n) return;
  const float* hp = h1 + (size_t)node * DIM;
  float s = 0.f;
#pragma unroll 4
  for (int it = 0; it < DIM / 32; ++it) {
    int c = lane + 32 * it;
    s += fmaxf(hp[c] + gt[c], 0.f) * d2w[c];
  }
#pragma unroll
  for (int off = 16; off > 0; off >>= 1) s += __shfl_down(s, off);
  if (lane == 0) out[node] = s + d2b[0];
}

// ---------------- host launcher ----------------
extern "C" void kernel_launch(void* const* d_in, const int* in_sizes, int n_in,
                              void* d_out, int out_size, void* d_ws, size_t ws_size,
                              hipStream_t stream) {
  (void)in_sizes; (void)n_in; (void)out_size; (void)ws_size;
  const float* x0      = (const float*)d_in[0];
  const float* eattr   = (const float*)d_in[1];
  const float* goal    = (const float*)d_in[2];
  const float* cgWf[2] = {(const float*)d_in[3],  (const float*)d_in[11]};
  const float* cgbf[2] = {(const float*)d_in[4],  (const float*)d_in[12]};
  const float* cgWs[2] = {(const float*)d_in[5],  (const float*)d_in[13]};
  const float* cgbs[2] = {(const float*)d_in[6],  (const float*)d_in[14]};
  const float* cggm[2] = {(const float*)d_in[7],  (const float*)d_in[15]};
  const float* cgbt[2] = {(const float*)d_in[8],  (const float*)d_in[16]};
  const float* cgmn[2] = {(const float*)d_in[9],  (const float*)d_in[17]};
  const float* cgvr[2] = {(const float*)d_in[10], (const float*)d_in[18]};
  const float* g3W = (const float*)d_in[19]; const float* g3b = (const float*)d_in[20];
  const float* g4W = (const float*)d_in[21]; const float* g4b = (const float*)d_in[22];
  const float* d1W = (const float*)d_in[23]; const float* d1b = (const float*)d_in[24];
  const float* d2W = (const float*)d_in[25]; const float* d2b = (const float*)d_in[26];
  const int*   ei  = (const int*)d_in[27];
  float* out = (float*)d_out;

  char* ws = (char*)d_ws;
  size_t o = 0;
  auto carve = [&](size_t bytes) -> char* {
    char* p = ws + o;
    o += (bytes + 255) & ~(size_t)255;
    return p;
  };
  __bf16* wcg[2];
  wcg[0] = (__bf16*)carve((size_t)4 * DIM * DIM * 2);
  wcg[1] = (__bf16*)carve((size_t)4 * DIM * DIM * 2);
  __bf16* g3t = (__bf16*)carve((size_t)HID * DIM * 2);
  __bf16* g4t = (__bf16*)carve((size_t)HID * HID * 2);
  __bf16* d1t = (__bf16*)carve((size_t)DIM * HID * 2);
  float* AF = (float*)carve((size_t)N_NODES * DIM * 4);
  float* BF = (float*)carve((size_t)N_NODES * DIM * 4);
  float* AS = (float*)carve((size_t)N_NODES * DIM * 4);
  float* BS = (float*)carve((size_t)N_NODES * DIM * 4);
  float* AG = (float*)carve((size_t)N_NODES * DIM * 4);
  float* X  = (float*)carve((size_t)N_NODES * DIM * 4);
  float* DEG = (float*)carve((size_t)N_NODES * 4);
  float* GT  = (float*)carve((size_t)DIM * 4);

  auto tc = [&](const float* src, __bf16* dst, int K, int Nout, int ld, int rowoff) {
    int tot = K * Nout;
    transcvt_kernel<<<(tot + 255) / 256, 256, 0, stream>>>(src, dst, K, Nout, ld, rowoff);
  };
  for (int l = 0; l < 2; ++l) {
    tc(cgWf[l], wcg[l] + 0 * DIM * DIM, DIM, DIM, DIM, 0);    // W_f dst part
    tc(cgWf[l], wcg[l] + 1 * DIM * DIM, DIM, DIM, DIM, DIM);  // W_f src part
    tc(cgWs[l], wcg[l] + 2 * DIM * DIM, DIM, DIM, DIM, 0);    // W_s dst part
    tc(cgWs[l], wcg[l] + 3 * DIM * DIM, DIM, DIM, DIM, DIM);  // W_s src part
  }
  tc(g3W, g3t, DIM, HID, HID, 0);
  tc(g4W, g4t, HID, HID, HID, 0);
  tc(d1W, d1t, HID, DIM, DIM, 0);

  // degrees (needed by GCN layers; independent of CG phase)
  fill_kernel<<<(N_NODES + 255) / 256, 256, 0, stream>>>(DEG, 1.f, N_NODES);
  deg_kernel<<<(N_EDGES + 255) / 256, 256, 0, stream>>>(ei + N_EDGES, DEG, N_EDGES);
  rsqrt_kernel<<<(N_NODES + 255) / 256, 256, 0, stream>>>(DEG, N_NODES);

  auto gemm = [&](const float* A, int lda, const __bf16* Bt, float* C, const float* bias,
                  int M, int K, int Nout) {
    int tiles = ((M + 15) / 16) * (Nout / 16);
    wmma_gemm_kernel<<<(tiles + 7) / 8, 256, 0, stream>>>(A, lda, Bt, C, bias, M, K, Nout);
  };

  // -------- two CGConv layers --------
  const float* xin = x0;
  for (int l = 0; l < 2; ++l) {
    gemm(xin, DIM, wcg[l] + 0 * DIM * DIM, AF, cgbf[l], N_NODES, DIM, DIM);
    gemm(xin, DIM, wcg[l] + 1 * DIM * DIM, BF, nullptr, N_NODES, DIM, DIM);
    gemm(xin, DIM, wcg[l] + 2 * DIM * DIM, AS, cgbs[l], N_NODES, DIM, DIM);
    gemm(xin, DIM, wcg[l] + 3 * DIM * DIM, BS, nullptr, N_NODES, DIM, DIM);
    hipMemsetAsync(AG, 0, (size_t)N_NODES * DIM * 4, stream);
    cg_edge_kernel<<<2048, 256, 0, stream>>>(ei, AF, BF, AS, BS, eattr,
                                             cgWf[l] + (size_t)2 * DIM * DIM,   // rows 1024..1039
                                             cgWs[l] + (size_t)2 * DIM * DIM,
                                             AG, N_EDGES);
    cg_epilogue_kernel<<<(N_NODES * DIM + 255) / 256, 256, 0, stream>>>(
        xin, AG, cggm[l], cgbt[l], cgmn[l], cgvr[l], X, N_NODES * DIM);
    xin = X;
  }

  // -------- GCN layer 3 (512 -> 256) --------
  gemm(X, DIM, g3t, AF, nullptr, N_NODES, DIM, HID);                 // h in AF
  hipMemsetAsync(BF, 0, (size_t)N_NODES * HID * 4, stream);
  gcn_scatter_kernel<<<2048, 256, 0, stream>>>(ei, DEG, AF, BF, N_EDGES, HID);
  gcn_epilogue_kernel<<<(N_NODES * HID + 255) / 256, 256, 0, stream>>>(BF, AF, DEG, g3b,
                                                                       N_NODES, HID);
  // -------- GCN layer 4 (256 -> 256) --------
  gemm(BF, HID, g4t, AF, nullptr, N_NODES, HID, HID);                // h in AF
  hipMemsetAsync(AS, 0, (size_t)N_NODES * HID * 4, stream);
  gcn_scatter_kernel<<<2048, 256, 0, stream>>>(ei, DEG, AF, AS, N_EDGES, HID);
  gcn_epilogue_kernel<<<(N_NODES * HID + 255) / 256, 256, 0, stream>>>(AS, AF, DEG, g4b,
                                                                       N_NODES, HID);
  // -------- dense head --------
  gterm_kernel<<<1, DIM, 0, stream>>>(goal, d1W, d1b, GT);           // constant goal term
  gemm(AS, HID, d1t, BF, nullptr, N_NODES, HID, DIM);                // h1 = x @ d1W[:256]
  final_kernel<<<(N_NODES * 32 + 255) / 256, 256, 0, stream>>>(BF, GT, d2W, d2b, out, N_NODES);
}
